// PG_26156350832878
// MI455X (gfx1250) — matrix operand order repaired and checked by reading, not compiled, not run on hardware
//
#include <hip/hip_runtime.h>

typedef _Float16 f16;
typedef _Float16 v16h  __attribute__((ext_vector_type(16)));
typedef float    v8f   __attribute__((ext_vector_type(8)));
typedef unsigned int u32x4 __attribute__((ext_vector_type(4)));

union Frag { v16h h; u32x4 q[2]; };

#define C_   128
#define HW_  196
#define CD_  147456   /* C*C*3*3 */
#define T_   30

#if defined(__HIP_DEVICE_COMPILE__) && __has_builtin(__builtin_amdgcn_tensor_load_to_lds) && __has_builtin(__builtin_amdgcn_s_wait_tensorcnt)
#define HAVE_TDM 1
#else
#define HAVE_TDM 0
#endif

#if HAVE_TDM
typedef unsigned int tdm_v4u __attribute__((ext_vector_type(4)));
typedef int          tdm_v8i __attribute__((ext_vector_type(8)));
typedef int          tdm_v4i __attribute__((ext_vector_type(4)));

// DMA a 2D tile (tile_d0 f16 elems per row x tile_d1 rows, row stride
// d0_stride elems) from global into LDS, padding 4 DWORDs after every
// 16 DWORDs stored -> matches our 40-half (80B) padded LDS rows when
// tile_d0 == 32.  Issued by one wave; EXEC is ignored by TDM.
__device__ __forceinline__ void tdm_load_2d_pad(
    unsigned lds_addr, const void* gaddr, unsigned tile_d0, unsigned tile_d1,
    unsigned tensor_d0, unsigned tensor_d1, unsigned long long d0_stride)
{
  unsigned long long ga = (unsigned long long)gaddr;
  tdm_v4u g0;
  g0[0] = 1u;                                        // count=1, user descriptor
  g0[1] = lds_addr;                                  // LDS byte address
  g0[2] = (unsigned)(ga & 0xffffffffull);            // global_addr[31:0]
  g0[3] = (unsigned)((ga >> 32) & 0x1ffffffull) | (2u << 30);  // addr[56:32], type=2
  tdm_v8i g1;
  unsigned w0 = (1u << 16)                           // data_size = 2 bytes
              | (1u << 20)                           // pad_enable
              | (3u << 22)                           // pad_interval: 16 DWORDs
              | (3u << 25);                          // pad_amount: 4 DWORDs
  g1[0] = (int)w0;                                   // workgroup_mask=0 (no cluster)
  g1[1] = (int)((tensor_d0 & 0xffffu) << 16);        // abar_addr=0 | tensor_dim0 lo
  g1[2] = (int)(((tensor_d0 >> 16) & 0xffffu) | ((tensor_d1 & 0xffffu) << 16));
  g1[3] = (int)(((tensor_d1 >> 16) & 0xffffu) | ((tile_d0 & 0xffffu) << 16));
  g1[4] = (int)(tile_d1 & 0xffffu);                  // tile_dim1 | tile_dim2=0
  g1[5] = (int)(unsigned)(d0_stride & 0xffffffffull);
  g1[6] = (int)(unsigned)((d0_stride >> 32) & 0xffffull);
  g1[7] = 0;
  tdm_v4i z4 = {0, 0, 0, 0};
  tdm_v8i z8 = {0, 0, 0, 0, 0, 0, 0, 0};
  __builtin_amdgcn_tensor_load_to_lds(g0, g1, z4, z4, z8, 0);
}
#endif

// token -> kind/uidx/bidx tables (from UNARY/BINARY lists in the reference)
__constant__ int c_kind[44] = {
  0,0,0,0,1, 3,3,3,3,3, 2,2,2,2,2,2,2,2,2,2,2,2,2,2,2,2, 3,3,3,
  2,2,2,2,2,2,2,2,2,2,2,2, 3, 2};
__constant__ int c_uidx[44] = {
  0,0,0,0,0, 0,0,0,0,0, 1,2,3,4,5,6,7,8,9,10,11,12,13,14,15,16, 0,0,0,
  17,18,19,20,21,22,23,24,25,26,27,28,29, 0, 30};
__constant__ int c_bidx[44] = {
  0,0,0,0,0, 0,1,2,3,4, 0,0,0,0,0,0,0,0,0,0,0,0,0,0,0,0, 5,6,7,
  0,0,0,0,0,0,0,0,0,0,0,0,0, 8, 0};

// ---------------------------------------------------------------------------
// Implicit-GEMM conv (3x3 pad 1, or 1x1) on a 14x14 grid via WMMA.
// A = weights [COUT][cin*TAPS] row-major (OIHW is exactly this, so the
// TDM tile is a plain strided 2D slab), streamed by the Tensor Data Mover
// with a depth-2 pipeline (DMA for chunk i+1 issued while chunk i computes).
// B = im2col tile (32 K x 32 pixels) built in LDS.  Block = 256 threads =
// 8 waves; each wave owns COUT/128 M-tiles of 16 rows x 2 N-tiles.
// LDS rows padded to 40 halves (80B) -> conflict-free ds_load_b128.
// ---------------------------------------------------------------------------
template<int COUT, int TAPS>
__device__ __forceinline__ void conv_core(
    const f16* __restrict__ W, const f16* __restrict__ inA,
    const f16* __restrict__ inB, int cin1, int cin,
    const float* __restrict__ bias, f16* __restrict__ outP,
    f16* __restrict__ out2)
{
  constexpr int MT = COUT / 128;
  __shared__ __align__(16) f16 As[2][COUT * 40];
  __shared__ __align__(16) f16 Bs[32 * 40];

  const int tid = threadIdx.x;
  const int wv  = tid >> 5;
  const int ln  = tid & 31;
  const int p0  = blockIdx.x * 32;
  const int Ktot = cin * TAPS;

  // hoisted per-thread B-fill state: kk constant, 4 pixels per thread
  const int kk    = tid & 31;
  const int pbase = tid >> 5;          // 0..7
  int py[4], px[4]; bool pv[4];
  #pragma unroll
  for (int i = 0; i < 4; ++i) {
    int pix = p0 + pbase + i * 8;
    pv[i] = pix < HW_;
    py[i] = pix / 14;
    px[i] = pix - py[i] * 14;
  }

  v8f acc[MT][2];
  v8f zv = {0.f,0.f,0.f,0.f,0.f,0.f,0.f,0.f};
  #pragma unroll
  for (int j = 0; j < MT; ++j) { acc[j][0] = zv; acc[j][1] = zv; }

  const int sel8 = (ln >> 4) * 8;      // A half-wave K sub-chunk
  const int koff = (ln >> 4) * 16;     // unused; B also indexes with sel8
  const int cl   = ln & 15;

  int buf = 0;
#if HAVE_TDM
  if (wv == 0) {                        // prologue DMA: chunk 0 -> buffer 0
    tdm_load_2d_pad((unsigned)(uintptr_t)(void*)As[0], W,
                    32u, (unsigned)COUT, (unsigned)Ktot, (unsigned)COUT,
                    (unsigned long long)Ktot);
  }
#endif
  for (int k0 = 0; k0 < Ktot; k0 += 32) {
#if HAVE_TDM
    if (wv == 0 && k0 + 32 < Ktot) {    // prefetch chunk i+1 into other buffer
      tdm_load_2d_pad((unsigned)(uintptr_t)(void*)As[buf ^ 1], W + k0 + 32,
                      32u, (unsigned)COUT, (unsigned)(Ktot - k0 - 32),
                      (unsigned)COUT, (unsigned long long)Ktot);
    }
#else
    #pragma unroll
    for (int c = tid; c < COUT * 4; c += 256) {
      int row  = c >> 2;
      int coff = (c & 3) * 8;
      const f16* gp = W + (size_t)row * Ktot + k0 + coff;
      __builtin_prefetch(gp + 32, 0, 1);
      *(u32x4*)&As[buf][row * 40 + coff] = *(const u32x4*)gp;
    }
#endif
    // ---- stage B (im2col gather): one K-decomposition per chunk ----
    {
      int kg = k0 + kk;
      int ci, ky = 0, kx = 0;
      if (TAPS == 9) {
        ci = kg / 9;
        int r = kg - ci * 9;
        ky = r / 3;
        kx = r - ky * 3;
      } else {
        ci = kg;
      }
      const f16* src = (ci < cin1) ? (inA + (size_t)ci * HW_)
                                   : (inB + (size_t)(ci - cin1) * HW_);
      #pragma unroll
      for (int i = 0; i < 4; ++i) {
        f16 v = (f16)0.f;
        if (pv[i]) {
          if (TAPS == 9) {
            int iy = py[i] + ky - 1;
            int ix = px[i] + kx - 1;
            if (iy >= 0 && iy < 14 && ix >= 0 && ix < 14) v = src[iy * 14 + ix];
          } else {
            v = src[py[i] * 14 + px[i]];
          }
        }
        Bs[(pbase + i * 8) * 40 + kk] = v;
      }
    }
#if HAVE_TDM
    if (wv == 0) {                      // in-order: <=1 pending => chunk i done
      if (k0 + 32 < Ktot) __builtin_amdgcn_s_wait_tensorcnt(1);
      else                __builtin_amdgcn_s_wait_tensorcnt(0);
    }
#endif
    __syncthreads();

    // ---- fragments per documented CDNA5 wave32 layouts + WMMA ----
    Frag b0, b1;
    b0.q[0] = *(const u32x4*)&Bs[cl * 40 + sel8];
    b0.q[1] = *(const u32x4*)&Bs[cl * 40 + 16 + sel8];
    b1.q[0] = *(const u32x4*)&Bs[(cl + 16) * 40 + sel8];
    b1.q[1] = *(const u32x4*)&Bs[(cl + 16) * 40 + 16 + sel8];
    #pragma unroll
    for (int j = 0; j < MT; ++j) {
      int r0 = (wv * MT + j) * 16 + cl;
      Frag a;
      a.q[0] = *(const u32x4*)&As[buf][r0 * 40 + sel8];
      a.q[1] = *(const u32x4*)&As[buf][r0 * 40 + 16 + sel8];
      acc[j][0] = __builtin_amdgcn_wmma_f32_16x16x32_f16(
          false, a.h, false, b0.h, (short)0, acc[j][0], false, false);
      acc[j][1] = __builtin_amdgcn_wmma_f32_16x16x32_f16(
          false, a.h, false, b1.h, (short)0, acc[j][1], false, false);
    }
    __syncthreads();
    buf ^= 1;
  }

  // ---- epilogue: bias + relu + f16 store (D: M = v + 8*(l/16), N = l%16) --
  #pragma unroll
  for (int j = 0; j < MT; ++j) {
    const int mbase = (wv * MT + j) * 16 + (ln >> 4) * 8;
    #pragma unroll
    for (int nt = 0; nt < 2; ++nt) {
      const int pix = p0 + nt * 16 + cl;
      if (pix < HW_) {
        #pragma unroll
        for (int vv = 0; vv < 8; ++vv) {
          int co = mbase + vv;
          float val = acc[j][nt][vv];
          if (bias) val += bias[co];
          val = fmaxf(val, 0.f);
          outP[(size_t)co * HW_ + pix] = (f16)val;
          if (out2) out2[(size_t)co * HW_ + pix] = (f16)val;
        }
      }
    }
  }
}

template<int COUT, int TAPS>
__global__ __launch_bounds__(256) void conv_fixed_kernel(
    const f16* __restrict__ W, const f16* __restrict__ in, int cin,
    const float* __restrict__ bias, f16* __restrict__ out,
    f16* __restrict__ out2)
{
  int b = blockIdx.y;
  const f16* inb = in + (size_t)b * cin * HW_;
  f16* ob = out + (size_t)b * COUT * HW_;
  f16* o2 = out2 ? out2 + (size_t)b * COUT * HW_ : nullptr;
  conv_core<COUT, TAPS>(W, inb, nullptr, cin, cin, bias, ob, o2);
}

// One program step, one conv stage. mode 0: first conv (unary/scene/binary),
// mode 1: second conv, mode 2: binary third conv. Blocks whose batch item
// doesn't need this stage exit immediately (halves work vs. reference).
__global__ __launch_bounds__(256) void step_conv_kernel(
    int mode, int t, const int* __restrict__ programs,
    const f16* __restrict__ mem_u16, const f16* __restrict__ mem_b16,
    const f16* __restrict__ feat16, f16* __restrict__ outB,
    f16* __restrict__ savedB, f16* __restrict__ hbuf, f16* __restrict__ h2buf)
{
  const int b    = blockIdx.y;
  const int tok  = programs[b * T_ + (T_ - 1 - t)];   // toks_rev order
  const int kind = c_kind[tok];
  if (kind == 0) return;
  if (mode == 2 && kind != 3) return;

  const size_t boff = (size_t)b * C_ * HW_;
  const int p0 = blockIdx.x * 32;

  const f16* W; const f16* inA; const f16* inB = nullptr;
  int cin = 128; f16* dst;

  if (mode == 0) {
    if (kind == 3) {                       // binary conv1: concat(out,saved)
      W = mem_b16 + (size_t)c_bidx[tok] * 4 * CD_;
      inA = outB + boff; inB = savedB + boff; cin = 256; dst = hbuf + boff;
    } else {
      if (kind == 1) {                     // scene: new_saved = old out
        for (int i = threadIdx.x; i < 128 * 32; i += 256) {
          int c = i >> 5, pp = i & 31, pix = p0 + pp;
          if (pix < HW_)
            savedB[boff + (size_t)c * HW_ + pix] =
                outB[boff + (size_t)c * HW_ + pix];
        }
      }
      W = mem_u16 + (size_t)c_uidx[tok] * 2 * CD_;
      inA = ((kind == 1) ? feat16 : outB) + boff; dst = hbuf + boff;
    }
  } else if (mode == 1) {
    if (kind == 3) {
      W = mem_b16 + (size_t)c_bidx[tok] * 4 * CD_ + 2 * CD_;
      inA = hbuf + boff; dst = h2buf + boff;
    } else {
      W = mem_u16 + (size_t)c_uidx[tok] * 2 * CD_ + CD_;
      inA = hbuf + boff; dst = outB + boff;
    }
  } else {
    W = mem_b16 + (size_t)c_bidx[tok] * 4 * CD_ + 3 * CD_;
    inA = h2buf + boff; dst = outB + boff;
  }
  conv_core<128, 9>(W, inA, inB, 128, cin, nullptr, dst, nullptr);
}

// FC1 as WMMA GEMM: M=1024 (weight rows), N=32 (batch), K=25088.
// Both operand tiles are strided 2D slabs: wave 0 streams A via TDM,
// wave 1 streams B via TDM, each with a depth-2 pipeline.
__global__ __launch_bounds__(256) void fc1_wmma_kernel(
    const f16* __restrict__ Wt, const f16* __restrict__ P,
    const float* __restrict__ bias, float* __restrict__ h1)
{
  __shared__ __align__(16) f16 As[2][128 * 40];
  __shared__ __align__(16) f16 Bs[2][32 * 40];
  const int tid = threadIdx.x, wv = tid >> 5, ln = tid & 31;
  const int row0 = blockIdx.x * 128;
  const int sel8 = (ln >> 4) * 8, koff = (ln >> 4) * 16, cl = ln & 15;
  v8f acc0 = {0.f,0.f,0.f,0.f,0.f,0.f,0.f,0.f};
  v8f acc1 = acc0;

  int buf = 0;
#if HAVE_TDM
  if (wv == 0) {
    tdm_load_2d_pad((unsigned)(uintptr_t)(void*)As[0],
                    Wt + (size_t)row0 * 25088,
                    32u, 128u, 25088u, 128u, 25088ull);
  } else if (wv == 1) {
    tdm_load_2d_pad((unsigned)(uintptr_t)(void*)Bs[0], P,
                    32u, 32u, 25088u, 32u, 25088ull);
  }
#endif
  for (int k0 = 0; k0 < 25088; k0 += 32) {
#if HAVE_TDM
    if (k0 + 32 < 25088) {
      if (wv == 0) {
        tdm_load_2d_pad((unsigned)(uintptr_t)(void*)As[buf ^ 1],
                        Wt + (size_t)row0 * 25088 + k0 + 32,
                        32u, 128u, (unsigned)(25088 - k0 - 32), 128u, 25088ull);
        __builtin_amdgcn_s_wait_tensorcnt(1);
      } else if (wv == 1) {
        tdm_load_2d_pad((unsigned)(uintptr_t)(void*)Bs[buf ^ 1], P + k0 + 32,
                        32u, 32u, (unsigned)(25088 - k0 - 32), 32u, 25088ull);
        __builtin_amdgcn_s_wait_tensorcnt(1);
      }
    } else if (wv <= 1) {
      __builtin_amdgcn_s_wait_tensorcnt(0);
    }
#else
    #pragma unroll
    for (int c = tid; c < 512; c += 256) {
      int row = c >> 2, coff = (c & 3) * 8;
      const f16* gp = Wt + (size_t)(row0 + row) * 25088 + k0 + coff;
      __builtin_prefetch(gp + 32, 0, 1);
      *(u32x4*)&As[buf][row * 40 + coff] = *(const u32x4*)gp;
    }
    if (tid < 128) {
      int col = tid >> 2, coff = (tid & 3) * 8;
      *(u32x4*)&Bs[buf][col * 40 + coff] =
          *(const u32x4*)(P + (size_t)col * 25088 + k0 + coff);
    }
#endif
    __syncthreads();
    Frag a, b0, b1;
    int r0 = wv * 16 + cl;
    a.q[0]  = *(const u32x4*)&As[buf][r0 * 40 + sel8];
    a.q[1]  = *(const u32x4*)&As[buf][r0 * 40 + 16 + sel8];
    b0.q[0] = *(const u32x4*)&Bs[buf][cl * 40 + sel8];
    b0.q[1] = *(const u32x4*)&Bs[buf][cl * 40 + 16 + sel8];
    b1.q[0] = *(const u32x4*)&Bs[buf][(cl + 16) * 40 + sel8];
    b1.q[1] = *(const u32x4*)&Bs[buf][(cl + 16) * 40 + 16 + sel8];
    acc0 = __builtin_amdgcn_wmma_f32_16x16x32_f16(false, a.h, false, b0.h,
                                                  (short)0, acc0, false, false);
    acc1 = __builtin_amdgcn_wmma_f32_16x16x32_f16(false, a.h, false, b1.h,
                                                  (short)0, acc1, false, false);
    __syncthreads();
    buf ^= 1;
  }
  #pragma unroll
  for (int vv = 0; vv < 8; ++vv) {
    int co = row0 + wv * 16 + (ln >> 4) * 8 + vv;
    float v0 = fmaxf(acc0[vv] + bias[co], 0.f);
    float v1 = fmaxf(acc1[vv] + bias[co], 0.f);
    h1[(size_t)cl * 1024 + co]        = v0;
    h1[(size_t)(cl + 16) * 1024 + co] = v1;
  }
}

__global__ void fc2_kernel(const float* __restrict__ h1,
                           const float* __restrict__ w,
                           const float* __restrict__ b, float* __restrict__ out)
{
  int i = blockIdx.x * 256 + threadIdx.x;         // 0..1023
  int bb = i >> 5, a = i & 31;
  float s = b[a];
  for (int k = 0; k < 1024; ++k) s += h1[bb * 1024 + k] * w[a * 1024 + k];
  out[bb * 32 + a] = s;
}

__global__ void pool_kernel(const f16* __restrict__ h, f16* __restrict__ P)
{
  int idx = blockIdx.x * 256 + threadIdx.x;       // B*512*49 = 802816
  if (idx >= 802816) return;
  int b = idx / 25088, rem = idx - b * 25088;
  int c = rem / 49, pr = rem - c * 49;
  int py = pr / 7, px = pr - py * 7;
  const f16* base = h + ((size_t)(b * 512 + c) * HW_) + (2 * py) * 14 + 2 * px;
  float m = (float)base[0];
  m = fmaxf(m, (float)base[1]);
  m = fmaxf(m, (float)base[14]);
  m = fmaxf(m, (float)base[15]);
  P[idx] = (f16)m;
}

__global__ void f32_to_f16_kernel(const float* __restrict__ s,
                                  f16* __restrict__ d, size_t n)
{
  size_t i = (size_t)blockIdx.x * blockDim.x + threadIdx.x;
  size_t st = (size_t)gridDim.x * blockDim.x;
  for (; i < n; i += st) d[i] = (f16)s[i];
}

__global__ void zero16_kernel(f16* __restrict__ d, size_t n)
{
  size_t i = (size_t)blockIdx.x * blockDim.x + threadIdx.x;
  size_t st = (size_t)gridDim.x * blockDim.x;
  for (; i < n; i += st) d[i] = (f16)0.f;
}

extern "C" void kernel_launch(void* const* d_in, const int* in_sizes, int n_in,
                              void* d_out, int out_size, void* d_ws, size_t ws_size,
                              hipStream_t stream)
{
  (void)in_sizes; (void)n_in; (void)out_size; (void)ws_size;
  const float* feats    = (const float*)d_in[0];
  const int*   programs = (const int*)d_in[1];
  const float* stem_w1  = (const float*)d_in[2];
  const float* stem_b1  = (const float*)d_in[3];
  const float* stem_w2  = (const float*)d_in[4];
  const float* stem_b2  = (const float*)d_in[5];
  const float* mem_u    = (const float*)d_in[6];
  const float* mem_b    = (const float*)d_in[7];
  const float* cls_w    = (const float*)d_in[8];
  const float* cls_b    = (const float*)d_in[9];
  const float* fc1_w    = (const float*)d_in[10];
  const float* fc1_b    = (const float*)d_in[11];
  const float* fc2_w    = (const float*)d_in[12];
  const float* fc2_b    = (const float*)d_in[13];
  float* out = (float*)d_out;

  char* wp = (char*)d_ws;
  auto take = [&](size_t bytes) {
    char* p = wp; wp += (bytes + 255) & ~(size_t)255; return p;
  };
  f16* feats16 = (f16*)take(6422528ull * 2);
  f16* w1_16   = (f16*)take(1179648ull * 2);
  f16* w2_16   = (f16*)take(147456ull * 2);
  f16* memu16  = (f16*)take(9142272ull * 2);
  f16* memb16  = (f16*)take(5308416ull * 2);
  f16* clsw16  = (f16*)take(65536ull * 2);
  f16* fc1w16  = (f16*)take(25690112ull * 2);
  f16* hstem   = (f16*)take(802816ull * 2);
  f16* feat16  = (f16*)take(802816ull * 2);
  f16* outB    = (f16*)take(802816ull * 2);
  f16* savedB  = (f16*)take(802816ull * 2);
  f16* hbuf    = (f16*)take(802816ull * 2);
  f16* h2buf   = (f16*)take(802816ull * 2);
  f16* h512    = (f16*)take(3211264ull * 2);
  f16* pooled  = (f16*)take(802816ull * 2);
  float* h1    = (float*)take(32768ull * 4);

  auto cvt = [&](const float* s, f16* d, size_t n) {
    int grid = (int)((n + 255) / 256); if (grid > 4096) grid = 4096;
    f32_to_f16_kernel<<<grid, 256, 0, stream>>>(s, d, n);
  };
  cvt(feats,   feats16, 6422528);
  cvt(stem_w1, w1_16,   1179648);
  cvt(stem_w2, w2_16,   147456);
  cvt(mem_u,   memu16,  9142272);
  cvt(mem_b,   memb16,  5308416);
  cvt(cls_w,   clsw16,  65536);
  cvt(fc1_w,   fc1w16,  25690112);
  zero16_kernel<<<256, 256, 0, stream>>>(savedB, (size_t)802816);

  dim3 grid(7, 32), blk(256);
  conv_fixed_kernel<128, 9><<<grid, blk, 0, stream>>>(
      w1_16, feats16, 1024, stem_b1, hstem, nullptr);
  conv_fixed_kernel<128, 9><<<grid, blk, 0, stream>>>(
      w2_16, hstem, 128, stem_b2, feat16, outB);      // feat and initial out

  for (int t = 0; t < T_; ++t)
    for (int m = 0; m < 3; ++m)
      step_conv_kernel<<<grid, blk, 0, stream>>>(
          m, t, programs, memu16, memb16, feat16, outB, savedB, hbuf, h2buf);

  conv_fixed_kernel<512, 1><<<grid, blk, 0, stream>>>(
      clsw16, outB, 128, cls_b, h512, nullptr);
  pool_kernel<<<3136, 256, 0, stream>>>(h512, pooled);
  fc1_wmma_kernel<<<8, 256, 0, stream>>>(fc1w16, pooled, fc1_b, h1);
  fc2_kernel<<<4, 256, 0, stream>>>(h1, fc2_w, fc2_b, out);
}
